// GCN_79963701117626
// MI455X (gfx1250) — matrix operand order repaired and checked by reading, not compiled, not run on hardware
//
#include <hip/hip_runtime.h>
#include <hip/hip_bf16.h>

// MI455X (gfx1250) implementation.
// Dominant work: 48 x (768x1024)@(1024x1024) aggregation GEMMs (~310 GFLOP)
// + conv1x1 GEMMs (~27 GFLOP). ~300 MB HBM traffic @ 23.3 TB/s -> compute
// bound, so all GEMMs run on v_wmma_f32_16x16x32_bf16 (f32 accumulate).
// A tiles are staged global->LDS with global_load_async_to_lds_b128
// (ASYNCcnt); B tiles are register-transposed into [n][k] LDS layout so
// both WMMA fragment loads are two 16B ds_load_b128 each.

typedef __attribute__((ext_vector_type(16))) __bf16 v16bf;
typedef __attribute__((ext_vector_type(8)))  float  v8f;
typedef unsigned short bfraw;   // raw bf16 storage (bit pattern only)

static __device__ __forceinline__ bfraw f2bf(float f) {
    unsigned u = __builtin_bit_cast(unsigned, f);
    u += 0x7FFFu + ((u >> 16) & 1u);          // round-to-nearest-even
    return (bfraw)(u >> 16);
}

static __device__ __forceinline__ void wait_async0() {
#if defined(__has_builtin) && __has_builtin(__builtin_amdgcn_s_wait_asynccnt)
    __builtin_amdgcn_s_wait_asynccnt(0);
#else
    asm volatile("s_wait_asynccnt 0" ::: "memory");
#endif
}

// ---------------------------------------------------------------------------
// float -> bf16 plain convert (weights)
// ---------------------------------------------------------------------------
__global__ void cvt_f32_bf16_kernel(const float* __restrict__ src,
                                    bfraw* __restrict__ dst, int n) {
    int i = blockIdx.x * 256 + threadIdx.x;
    if (i < n) dst[i] = f2bf(src[i]);
}

// ---------------------------------------------------------------------------
// x (N,64,1024,12) f32 -> Xc bf16 (same layout, conv B-operand)
//                       -> XA bf16 [n][(c*12+l)][v]  (aggregation A-operand)
// ---------------------------------------------------------------------------
__global__ void cvt_x_kernel(const float* __restrict__ x,
                             bfraw* __restrict__ Xc,
                             bfraw* __restrict__ XA) {
    size_t i = (size_t)blockIdx.x * 256 + threadIdx.x;
    const size_t total = (size_t)16 * 64 * 1024 * 12;
    if (i >= total) return;
    bfraw b = f2bf(x[i]);
    Xc[i] = b;
    int    l = (int)(i % 12);
    size_t t = i / 12;
    int    v = (int)(t % 1024); t /= 1024;
    int    c = (int)(t % 64);
    int    n = (int)(t / 64);
    XA[((size_t)n * 768 + c * 12 + l) * 1024 + v] = b;
}

// ---------------------------------------------------------------------------
// Top-k + softmax adjacency builder. One wave32 per row of a (V=1024).
// Iterative max-selection (40 rounds) gives thresholds for k=10/20/40;
// then one pass builds the 4 softmaxed bf16 matrices (k10,k20,k40,full).
// ---------------------------------------------------------------------------
__global__ __launch_bounds__(32) void topk_softmax_kernel(
        const float* __restrict__ arow_base,   // support + (n*3+i)*V*V
        bfraw* __restrict__ Abuf) {            // [4][1024][1024] bf16
    const int row  = blockIdx.x;
    const int lane = threadIdx.x;
    const float* rp = arow_base + (size_t)row * 1024;

    float vals[32], work[32];
#pragma unroll
    for (int j = 0; j < 32; ++j) { vals[j] = rp[lane + 32 * j]; work[j] = vals[j]; }

    const float NEG = -3.4e38f;
    float t10 = NEG, t20 = NEG, t40 = NEG, rowmax = NEG;
    for (int it = 0; it < 40; ++it) {
        float m = NEG;
#pragma unroll
        for (int j = 0; j < 32; ++j) m = fmaxf(m, work[j]);
#pragma unroll
        for (int off = 16; off > 0; off >>= 1)
            m = fmaxf(m, __shfl_xor(m, off, 32));
        if (it == 0)  rowmax = m;
        if (it == 9)  t10 = m;
        if (it == 19) t20 = m;
        if (it == 39) t40 = m;
#pragma unroll
        for (int j = 0; j < 32; ++j) if (work[j] == m) work[j] = NEG;
    }

    float s10 = 0.f, s20 = 0.f, s40 = 0.f, sf = 0.f;
#pragma unroll
    for (int j = 0; j < 32; ++j) {
        float e = __expf(vals[j] - rowmax);
        sf += e;
        if (vals[j] >= t40) s40 += e;
        if (vals[j] >= t20) s20 += e;
        if (vals[j] >= t10) s10 += e;
    }
#pragma unroll
    for (int off = 16; off > 0; off >>= 1) {
        s10 += __shfl_xor(s10, off, 32);
        s20 += __shfl_xor(s20, off, 32);
        s40 += __shfl_xor(s40, off, 32);
        sf  += __shfl_xor(sf,  off, 32);
    }
    const float r10 = 1.f / s10, r20 = 1.f / s20, r40 = 1.f / s40, rf = 1.f / sf;

#pragma unroll
    for (int j = 0; j < 32; ++j) {
        float e = __expf(vals[j] - rowmax);
        size_t base = (size_t)row * 1024 + (lane + 32 * j);
        Abuf[0u * 1048576 + base] = f2bf(vals[j] >= t10 ? e * r10 : 0.f);
        Abuf[1u * 1048576 + base] = f2bf(vals[j] >= t20 ? e * r20 : 0.f);
        Abuf[2u * 1048576 + base] = f2bf(vals[j] >= t40 ? e * r40 : 0.f);
        Abuf[3u * 1048576 + base] = f2bf(e * rf);
    }
}

// ---------------------------------------------------------------------------
// Generic bf16 WMMA GEMM:  C(MxN) = A(MxK,row-major,lda) @ B(KxN,row-major,ldb)
// Block: 256 threads (8 waves as 4(M) x 2(N)), tile BM=SM*64 x BN=128, BK=32.
// A tile: async-copied global->LDS (global_load_async_to_lds_b128, ASYNCcnt).
// B tile: 8x2 column-pair patches, register transpose, 2x b128 into [n][k].
// Both WMMA fragments then load as two 16B ds_load_b128 matching the CDNA5
// 16-bit A (e0..7=K half*8+j, e8..15=K 16+half*8+j) and B (same K order as A)
// lane layouts.
// mode 0: H scatter  o[(z*64 + gm/12)*12288 + gn*12 + gm%12] = bf16(v)
// mode 1: conv store o[(chan_base+gm)*N + gn] = bf16(v + bias[gm]), o += z*out_zs
// mode 2: f32 store  o[gm*N + gn] = v + bias[gm],                   o += z*out_zs
// ---------------------------------------------------------------------------
#define BN 128
#define BK 32

template <int SM>   // BM = SM*64, M-subtiles per wave = SM
__global__ __launch_bounds__(256) void gemm_bf16_kernel(
        const bfraw* __restrict__ A, const bfraw* __restrict__ B,
        const float* __restrict__ bias, void* __restrict__ outp,
        int M, int N, int K, int lda, int ldb,
        long long b_zstride, long long out_zstride,
        int chan_base, int mode) {
    __shared__ bfraw sA[SM * 64 * BK];   // [m][k]
    __shared__ bfraw sB[BN * BK];        // [n][k] (transposed)

    const int z  = blockIdx.z;
    B += (long long)z * b_zstride;
    const int m0 = blockIdx.y * (SM * 64);
    const int n0 = blockIdx.x * BN;
    const int tid  = threadIdx.x;
    const int wave = tid >> 5, lane = tid & 31;
    const int wm = (wave & 3) * (SM * 16);   // 4 waves along M
    const int wn = (wave >> 2) * 64;         // 2 waves along N
    const int r16 = lane & 15, half = lane >> 4;

    // B-tile cooperative-load mapping: 64 column pairs x 4 k-groups
    const int n2 = tid & 63;        // column pair -> cols 2*n2, 2*n2+1
    const int kg = tid >> 6;        // k-group: rows kg*8 .. kg*8+7

    v8f acc[SM][4];
#pragma unroll
    for (int i = 0; i < SM; ++i)
#pragma unroll
        for (int j = 0; j < 4; ++j)
#pragma unroll
            for (int e = 0; e < 8; ++e) acc[i][j][e] = 0.f;

    for (int k0 = 0; k0 < K; k0 += BK) {
        // ---- A tile: BMx32, async global->LDS, 16B per lane per issue ----
#pragma unroll
        for (int s = 0; s < SM; ++s) {
            int id  = tid + 256 * s;
            int row = id >> 2;
            int cg  = (id & 3) * 8;
            unsigned ldsoff = (unsigned)(size_t)(&sA[row * BK + cg]);
            const bfraw* gp = A + (size_t)(m0 + row) * lda + k0 + cg;
            asm volatile("global_load_async_to_lds_b128 %0, %1, off"
                         :: "v"(ldsoff), "v"((unsigned long long)(size_t)gp)
                         : "memory");
        }
        // ---- B tile: 32x128, 8x2 patch per thread, register transpose ----
        {
            const bfraw* gb = B + (size_t)(k0 + kg * 8) * ldb + n0 + n2 * 2;
            unsigned u[8];
#pragma unroll
            for (int r = 0; r < 8; ++r)
                u[r] = *(const unsigned*)(gb + (size_t)r * ldb);
            uint4 q0, q1;
            q0.x = (u[0] & 0xFFFFu) | (u[1] << 16);
            q0.y = (u[2] & 0xFFFFu) | (u[3] << 16);
            q0.z = (u[4] & 0xFFFFu) | (u[5] << 16);
            q0.w = (u[6] & 0xFFFFu) | (u[7] << 16);
            q1.x = (u[0] >> 16) | (u[1] & 0xFFFF0000u);
            q1.y = (u[2] >> 16) | (u[3] & 0xFFFF0000u);
            q1.z = (u[4] >> 16) | (u[5] & 0xFFFF0000u);
            q1.w = (u[6] >> 16) | (u[7] & 0xFFFF0000u);
            *(uint4*)&sB[(n2 * 2 + 0) * BK + kg * 8] = q0;
            *(uint4*)&sB[(n2 * 2 + 1) * BK + kg * 8] = q1;
        }
        wait_async0();          // A tile resident in LDS
        __syncthreads();        // B tile visible to all waves

        v16bf af[SM], bfr[4];
#pragma unroll
        for (int i = 0; i < SM; ++i) {
            int r = wm + i * 16 + r16;
            union { uint4 q[2]; v16bf v; } ua;
            ua.q[0] = *(const uint4*)&sA[r * BK + half * 8];
            ua.q[1] = *(const uint4*)&sA[r * BK + 16 + half * 8];
            af[i] = ua.v;
        }
#pragma unroll
        for (int j = 0; j < 4; ++j) {
            int c = wn + j * 16 + r16;
            union { uint4 q[2]; v16bf v; } ub;
            ub.q[0] = *(const uint4*)&sB[c * BK + half * 8];
            ub.q[1] = *(const uint4*)&sB[c * BK + 16 + half * 8];
            bfr[j] = ub.v;
        }
#pragma unroll
        for (int i = 0; i < SM; ++i)
#pragma unroll
            for (int j = 0; j < 4; ++j)
                acc[i][j] = __builtin_amdgcn_wmma_f32_16x16x32_bf16(
                    false, af[i], false, bfr[j], (short)0, acc[i][j], false, false);
        __syncthreads();
    }

    // store: C tile element (vgpr e): M = e + 8*half, N = lane&15
#pragma unroll
    for (int i = 0; i < SM; ++i)
#pragma unroll
        for (int j = 0; j < 4; ++j)
#pragma unroll
            for (int e = 0; e < 8; ++e) {
                int gm = m0 + wm + i * 16 + 8 * half + e;
                int gn = n0 + wn + j * 16 + r16;
                float v = acc[i][j][e];
                if (mode == 0) {
                    int c = gm / 12, l = gm - c * 12;
                    bfraw* o = (bfraw*)outp;
                    o[(size_t)(z * 64 + c) * 12288 + (size_t)gn * 12 + l] = f2bf(v);
                } else if (mode == 1) {
                    bfraw* o = (bfraw*)outp + (long long)z * out_zstride;
                    o[(size_t)(chan_base + gm) * (size_t)N + gn] = f2bf(v + bias[gm]);
                } else {
                    float* o = (float*)outp + (long long)z * out_zstride;
                    o[(size_t)gm * (size_t)N + gn] = v + bias[gm];
                }
            }
}

// ---------------------------------------------------------------------------
// host side
// ---------------------------------------------------------------------------
static inline size_t al256(size_t x) { return (x + 255) & ~(size_t)255; }

extern "C" void kernel_launch(void* const* d_in, const int* in_sizes, int n_in,
                              void* d_out, int out_size, void* d_ws, size_t ws_size,
                              hipStream_t stream) {
    (void)in_sizes; (void)n_in; (void)out_size; (void)ws_size;
    const float* x       = (const float*)d_in[0];  // (16,64,1024,12)
    const float* support = (const float*)d_in[1];  // (16,3,1024,1024)
    const float* W0      = (const float*)d_in[2];  // (64,64)
    const float* b0      = (const float*)d_in[3];  // (64,)
    const float* Ws      = (const float*)d_in[4];  // (3,64,256)
    const float* bs      = (const float*)d_in[5];  // (3,64)
    const float* Wf      = (const float*)d_in[6];  // (64,256)
    const float* bfv     = (const float*)d_in[7];  // (64,)
    float* out = (float*)d_out;                    // (16,64,1024,12)

    char* ws = (char*)d_ws;
    size_t off = 0;
    bfraw* W0b = (bfraw*)(ws + off); off += al256((size_t)4096 * 2);
    bfraw* Wsb = (bfraw*)(ws + off); off += al256((size_t)49152 * 2);
    bfraw* Wfb = (bfraw*)(ws + off); off += al256((size_t)16384 * 2);
    bfraw* Xc  = (bfraw*)(ws + off); off += al256((size_t)12582912 * 2);
    bfraw* XA  = (bfraw*)(ws + off); off += al256((size_t)12582912 * 2);
    bfraw* Ab  = (bfraw*)(ws + off); off += al256((size_t)4 * 1048576 * 2);
    bfraw* Hb  = (bfraw*)(ws + off); off += al256((size_t)256 * 12288 * 2);
    bfraw* OUT = (bfraw*)(ws + off); off += al256((size_t)16 * 256 * 12288 * 2);

    // weight + x conversion
    cvt_f32_bf16_kernel<<<(4096 + 255) / 256, 256, 0, stream>>>(W0, W0b, 4096);
    cvt_f32_bf16_kernel<<<(49152 + 255) / 256, 256, 0, stream>>>(Ws, Wsb, 49152);
    cvt_f32_bf16_kernel<<<(16384 + 255) / 256, 256, 0, stream>>>(Wf, Wfb, 16384);
    cvt_x_kernel<<<(12582912 + 255) / 256, 256, 0, stream>>>(x, Xc, XA);

    // out0 = W0 @ x + b0 -> OUT channels [0,64), all n (z=batch)
    gemm_bf16_kernel<1><<<dim3(96, 1, 16), 256, 0, stream>>>(
        W0b, Xc, b0, OUT, 64, 12288, 64, 64, 12288,
        (long long)64 * 12288, (long long)256 * 12288, 0, 1);

    for (int i = 0; i < 3; ++i) {
        for (int n = 0; n < 16; ++n) {
            // build the 4 softmaxed adjacency matrices for (n, i)
            topk_softmax_kernel<<<1024, 32, 0, stream>>>(
                support + ((size_t)n * 3 + i) * 1048576, Ab);
            // aggregation: H[p*64+c][w*12+l] = sum_v XA[n][(c,l)][v] * A_p[v][w]
            gemm_bf16_kernel<2><<<dim3(8, 6, 4), 256, 0, stream>>>(
                XA + (size_t)n * 786432, Ab, nullptr, Hb,
                768, 1024, 1024, 1024, 1024,
                (long long)1048576, 0, 0, 0);
            // conv: Ws[i] @ H + bs[i] -> OUT channels [(i+1)*64, (i+2)*64)
            gemm_bf16_kernel<1><<<dim3(96, 1, 1), 256, 0, stream>>>(
                Wsb + (size_t)i * 16384, Hb, bs + i * 64,
                OUT + (size_t)n * 3145728,
                64, 12288, 256, 256, 12288,
                0, 0, (i + 1) * 64, 1);
        }
    }

    // final: Wf @ OUT + bf -> d_out (f32), all n (z=batch)
    gemm_bf16_kernel<1><<<dim3(96, 1, 16), 256, 0, stream>>>(
        Wfb, OUT, bfv, out, 64, 12288, 256, 256, 12288,
        (long long)256 * 12288, (long long)64 * 12288, 0, 2);
}
